// MultiHeadAttention_26431228740340
// MI455X (gfx1250) — compile-verified, matrix operand order checked by reading, not run on hardware
//
#include <hip/hip_runtime.h>
#include <hip/hip_bf16.h>
#include <cstdint>

typedef _Float16 f16;
typedef _Float16 v16h __attribute__((ext_vector_type(16)));
typedef float    v8f  __attribute__((ext_vector_type(8)));
typedef unsigned int u32;
typedef u32 u32x4 __attribute__((ext_vector_type(4)));
typedef u32 u32x8 __attribute__((ext_vector_type(8)));

union Frag16 { u32x4 u[2]; v16h h; };

#define HIDDEN 768
#define NTOK   4096          // 4 * 1024 tokens
#define COEFF  8             // GRID_SIZE + SPLINE_ORDER
#define KCAT   6912          // 768 + 768*8, divisible by 32
#define SEQ    1024
#define HEADS  12
#define HDIM   64

// ---------------------------------------------------------------------------
// TDM: issue a 2-D tensor_load_to_lds.  Tile = tile_y rows x tile_x elements
// (f16, data_size code 1), row stride = stride_elems, dest LDS row-major
// [tile_y][tile_x] contiguous.  Descriptor per CDNA5 ISA D# groups 0/1.
// ---------------------------------------------------------------------------
__device__ __forceinline__ void tdm_load_2d_f16(
    u32 lds_addr, const f16* gptr,
    u32 tile_x, u32 tile_y, u32 tensor_x, u32 tensor_y, u32 stride_elems)
{
    unsigned long long ga = (unsigned long long)(uintptr_t)gptr;
    u32x4 g0;
    g0[0] = 1u;                                           // count=1 (valid), user mode
    g0[1] = lds_addr;                                     // LDS byte address
    g0[2] = (u32)ga;                                      // global_addr[31:0]
    g0[3] = (u32)((ga >> 32) & 0x1FFFFFFu) | (2u << 30);  // addr[56:32] | type=2
    u32x8 g1;
    g1[0] = (1u << 16);                                   // data_size=1 (2 bytes)
    g1[1] = (tensor_x & 0xFFFFu) << 16;                   // tensor_dim0[15:0]
    g1[2] = (tensor_x >> 16) | ((tensor_y & 0xFFFFu) << 16); // td0[31:16] | td1[15:0]
    g1[3] = (tensor_y >> 16) | (tile_x << 16);            // td1[31:16] | tile_dim0
    g1[4] = tile_y & 0xFFFFu;                             // tile_dim1 | tile_dim2=0
    g1[5] = stride_elems;                                 // tensor_dim0_stride[31:0]
    g1[6] = 0;                                            // stride[47:32] | d1_stride[15:0]
    g1[7] = 0;                                            // tensor_dim1_stride[47:16]
    u32x4 gz = {0u, 0u, 0u, 0u};                          // groups 2/3: zero (2-D)
    asm volatile("tensor_load_to_lds %0, %1, %2, %3"
                 :: "s"(g0), "s"(g1), "s"(gz), "s"(gz)
                 : "memory");
}

// ---------------------------------------------------------------------------
// Kernel 1: build concatenated f16 weights  W[o][0..767]=base_w, [768..]=spline*scaler
// ---------------------------------------------------------------------------
__global__ void kan_prep_weights(
    const float* __restrict__ qbw, const float* __restrict__ qsw, const float* __restrict__ qsc,
    const float* __restrict__ kbw, const float* __restrict__ ksw, const float* __restrict__ ksc,
    const float* __restrict__ vbw, const float* __restrict__ vsw, const float* __restrict__ vsc,
    f16* __restrict__ Wq, f16* __restrict__ Wk, f16* __restrict__ Wv)
{
    int idx = blockIdx.x * blockDim.x + threadIdx.x;
    if (idx >= HIDDEN * HIDDEN) return;
    int p = blockIdx.y;
    const float* bw = (p == 0) ? qbw : (p == 1) ? kbw : vbw;
    const float* sw = (p == 0) ? qsw : (p == 1) ? ksw : vsw;
    const float* sc = (p == 0) ? qsc : (p == 1) ? ksc : vsc;
    f16* W          = (p == 0) ? Wq  : (p == 1) ? Wk  : Wv;

    int o = idx / HIDDEN, i = idx % HIDDEN;
    W[(size_t)o * KCAT + i] = (f16)bw[idx];
    float s = sc[idx];
#pragma unroll
    for (int c = 0; c < COEFF; c++)
        W[(size_t)o * KCAT + HIDDEN + i * COEFF + c] = (f16)(sw[(size_t)idx * COEFF + c] * s);
}

__global__ void convert_w_f16(const float* __restrict__ src, f16* __restrict__ dst, int n)
{
    int idx = blockIdx.x * blockDim.x + threadIdx.x;
    if (idx < n) dst[idx] = (f16)src[idx];
}

// ---------------------------------------------------------------------------
// Kernel 2: build concatenated activations A[n][0..767]=silu(x), [768..]=bspline bases
// Cox-de Boor on uniform grid: grid[j] = -1 + (j-3)*0.4, j = 0..11
// ---------------------------------------------------------------------------
__global__ void kan_prep_acts(const float* __restrict__ x, f16* __restrict__ A)
{
    int idx = blockIdx.x * blockDim.x + threadIdx.x;
    if (idx >= NTOK * HIDDEN) return;
    int n = idx / HIDDEN, i = idx % HIDDEN;
    float xv  = x[idx];
    float sil = xv / (1.0f + __expf(-xv));
    A[(size_t)n * KCAT + i] = (f16)sil;

    const float h = 0.4f;
    float bas[11];
#pragma unroll
    for (int j = 0; j < 11; j++) {
        float g0 = -1.0f + (float)(j - 3) * h;
        bas[j] = (xv >= g0 && xv < g0 + h) ? 1.0f : 0.0f;
    }
#pragma unroll
    for (int k = 1; k <= 3; k++) {
        float inv = 1.0f / ((float)k * h);
#pragma unroll
        for (int j = 0; j < 11; j++) {
            if (j < 11 - k) {
                float gj = -1.0f + (float)(j - 3) * h;       // grid[j]
                float gk = -1.0f + (float)(j + k - 2) * h;   // grid[j+k+1]
                bas[j] = (xv - gj) * inv * bas[j] + (gk - xv) * inv * bas[j + 1];
            }
        }
    }
#pragma unroll
    for (int c = 0; c < COEFF; c++)
        A[(size_t)n * KCAT + HIDDEN + i * COEFF + c] = (f16)bas[c];
}

// ---------------------------------------------------------------------------
// Kernel 3: NT GEMM  C[M,N](f32) = A[M,K](f16) @ W[N,K](f16)^T  (+ bias)
// Block 256 thr = 8 waves; block tile 64x256; wave tile 32x64 (8 WMMA / K-step).
// Tiles streamed into double-buffered LDS by the Tensor Data Mover (wave 0
// issues descriptors, TENSORcnt-synchronized), so compute waves never touch
// the copy path.
// ---------------------------------------------------------------------------
__global__ __launch_bounds__(256) void gemm_nt_f16f32(
    const f16* __restrict__ A, const f16* __restrict__ W,
    float* __restrict__ C, const float* __restrict__ bias,
    int M, int N, int K)
{
    __shared__ alignas(16) f16 sA[2][64 * 32];
    __shared__ alignas(16) f16 sB[2][256 * 32];

    int t    = threadIdx.x;
    int lane = t & 31, w = t >> 5;
    int half = lane >> 4, m = lane & 15;
    int wm = w & 1, wn = w >> 1;
    int gm0 = blockIdx.x * 64;
    int gn0 = blockIdx.y * 256;

    u32 ldsA[2] = { (u32)(uintptr_t)&sA[0][0], (u32)(uintptr_t)&sA[1][0] };
    u32 ldsB[2] = { (u32)(uintptr_t)&sB[0][0], (u32)(uintptr_t)&sB[1][0] };

    v8f z = {};
    v8f acc[2][4];
#pragma unroll
    for (int mi = 0; mi < 2; mi++)
#pragma unroll
        for (int nj = 0; nj < 4; nj++) acc[mi][nj] = z;

    // prologue: DMA first tile pair into buffer 0
    if (w == 0) {
        tdm_load_2d_f16(ldsA[0], A + (size_t)gm0 * K, 32, 64,  (u32)K, (u32)M, (u32)K);
        tdm_load_2d_f16(ldsB[0], W + (size_t)gn0 * K, 32, 256, (u32)K, (u32)N, (u32)K);
    }

    for (int k0 = 0; k0 < K; k0 += 32) {
        int buf = (k0 >> 5) & 1;
        if (w == 0) {
            if (k0 + 32 < K) {           // prefetch next pair into other buffer
                tdm_load_2d_f16(ldsA[buf ^ 1], A + (size_t)gm0 * K + k0 + 32,
                                32, 64,  (u32)K, (u32)M, (u32)K);
                tdm_load_2d_f16(ldsB[buf ^ 1], W + (size_t)gn0 * K + k0 + 32,
                                32, 256, (u32)K, (u32)N, (u32)K);
                __builtin_amdgcn_s_wait_tensorcnt(2);   // current pair complete
            } else {
                __builtin_amdgcn_s_wait_tensorcnt(0);
            }
        }
        __syncthreads();                 // release compute waves on current buffer

        Frag16 af[2], bf[4];
#pragma unroll
        for (int mi = 0; mi < 2; mi++) {
            const f16* p = &sA[buf][(wm * 32 + mi * 16 + m) * 32 + half * 8];
            af[mi].u[0] = *(const u32x4*)p;
            af[mi].u[1] = *(const u32x4*)(p + 16);
        }
#pragma unroll
        for (int nj = 0; nj < 4; nj++) {
            const f16* p = &sB[buf][(wn * 64 + nj * 16 + m) * 32 + half * 8];
            bf[nj].u[0] = *(const u32x4*)p;
            bf[nj].u[1] = *(const u32x4*)(p + 16);
        }
#pragma unroll
        for (int mi = 0; mi < 2; mi++)
#pragma unroll
            for (int nj = 0; nj < 4; nj++)
                acc[mi][nj] = __builtin_amdgcn_wmma_f32_16x16x32_f16(
                    false, af[mi].h, false, bf[nj].h, (short)0, acc[mi][nj], false, false);

        __syncthreads();                 // reads done before TDM overwrites buffer
    }

#pragma unroll
    for (int mi = 0; mi < 2; mi++)
#pragma unroll
        for (int nj = 0; nj < 4; nj++)
#pragma unroll
            for (int r = 0; r < 8; r++) {
                int row = gm0 + wm * 32 + mi * 16 + r + 8 * half;
                int col = gn0 + wn * 64 + nj * 16 + m;
                float v = acc[mi][nj][r];
                if (bias) v += bias[col];
                C[(size_t)row * N + col] = v;
            }
}

// ---------------------------------------------------------------------------
// Kernel 4: RoPE on q,k (f32) + convert q,k,v to f16 in [b*12+h][s][d] layout
// ---------------------------------------------------------------------------
__global__ void rope_convert(const float* __restrict__ q, const float* __restrict__ k,
                             const float* __restrict__ v,
                             f16* __restrict__ qh, f16* __restrict__ kh, f16* __restrict__ vh)
{
    int idx = blockIdx.x * blockDim.x + threadIdx.x;
    if (idx >= NTOK * HIDDEN) return;
    int n = idx / HIDDEN, j = idx % HIDDEN;
    int hh = j >> 6, d = j & 63;
    int b = n >> 10, s = n & 1023;
    int d2 = d & 31;
    float invf = __expf(-(float)d2 * (9.210340371976184f / 32.0f)); // theta^(-2i/64)
    float fr = (float)s * invf;
    float cs = __cosf(fr), sn = __sinf(fr);

    int base = n * HIDDEN + hh * 64;
    float qv = q[idx], kv = k[idx];
    float qr = (d < 32) ? -q[base + d + 32] : q[base + d - 32];
    float kr = (d < 32) ? -k[base + d + 32] : k[base + d - 32];

    size_t dst = ((size_t)(b * HEADS + hh) * SEQ + s) * HDIM + d;
    qh[dst] = (f16)(qv * cs + qr * sn);
    kh[dst] = (f16)(kv * cs + kr * sn);
    vh[dst] = (f16)v[idx];
}

// ---------------------------------------------------------------------------
// Kernel 5: flash attention. Grid = 48 head-batches x 16 q-tiles (64 rows).
// Block 128 thr = 4 waves; each wave owns 16 q rows, online softmax, WMMA.
// Output written as f16 [b][s][h*64+d] (i.e. 4096x768 row-major).
// ---------------------------------------------------------------------------
__global__ __launch_bounds__(128) void flash_attn(
    const f16* __restrict__ qh, const f16* __restrict__ kh, const f16* __restrict__ vh,
    f16* __restrict__ attn)
{
    __shared__ alignas(16) f16 sK[32 * 64];
    __shared__ alignas(16) f16 sVt[64 * 32];
    __shared__ alignas(16) f16 sP[4][16 * 32];

    int t    = threadIdx.x;
    int lane = t & 31, w = t >> 5;
    int half = lane >> 4, m = lane & 15;
    int bh = blockIdx.x >> 4;     // 0..47
    int qt = blockIdx.x & 15;
    int b = bh / HEADS, hh = bh % HEADS;
    size_t hb = (size_t)bh * SEQ * HDIM;

    Frag16 qf[2];
    {
        int row = qt * 64 + w * 16 + m;
        const f16* p = &qh[hb + (size_t)row * HDIM];
#pragma unroll
        for (int fi = 0; fi < 2; fi++) {
            qf[fi].u[0] = *(const u32x4*)(p + fi * 32 + half * 8);
            qf[fi].u[1] = *(const u32x4*)(p + fi * 32 + 16 + half * 8);
        }
    }

    float mrow[8], lrow[8];
    v8f z = {};
    v8f o[4];
#pragma unroll
    for (int r = 0; r < 8; r++) { mrow[r] = -1e30f; lrow[r] = 0.0f; }
#pragma unroll
    for (int j = 0; j < 4; j++) o[j] = z;

    for (int kt = 0; kt < SEQ / 32; kt++) {
        __syncthreads();
        {
            int off = t * 16;                           // stage K tile (row-major)
            *(u32x4*)&sK[off]     = *(const u32x4*)&kh[hb + (size_t)kt * 2048 + off];
            *(u32x4*)&sK[off + 8] = *(const u32x4*)&kh[hb + (size_t)kt * 2048 + off + 8];
#pragma unroll
            for (int e = 0; e < 16; e++) {              // stage V transposed [d][key]
                int idx = t * 16 + e;
                int key = idx >> 6, d = idx & 63;
                sVt[d * 32 + key] = vh[hb + (size_t)kt * 2048 + idx];
            }
        }
        __syncthreads();

        // scores: S[16 x 32] = Q(16x64) @ K_tile^T
        v8f s0 = z, s1 = z;
#pragma unroll
        for (int fi = 0; fi < 2; fi++) {
            Frag16 b0, b1;
            const f16* p0 = &sK[(0 * 16 + m) * 64 + fi * 32 + half * 8];
            const f16* p1 = &sK[(1 * 16 + m) * 64 + fi * 32 + half * 8];
            b0.u[0] = *(const u32x4*)p0; b0.u[1] = *(const u32x4*)(p0 + 16);
            b1.u[0] = *(const u32x4*)p1; b1.u[1] = *(const u32x4*)(p1 + 16);
            s0 = __builtin_amdgcn_wmma_f32_16x16x32_f16(false, qf[fi].h, false, b0.h, (short)0, s0, false, false);
            s1 = __builtin_amdgcn_wmma_f32_16x16x32_f16(false, qf[fi].h, false, b1.h, (short)0, s1, false, false);
        }

        // online softmax, rows live across 16-lane halves
#pragma unroll
        for (int r = 0; r < 8; r++) {
            float a0 = s0[r] * 0.125f, a1 = s1[r] * 0.125f;   // 1/sqrt(64)
            float mx = fmaxf(a0, a1);
#pragma unroll
            for (int off = 8; off >= 1; off >>= 1) mx = fmaxf(mx, __shfl_xor(mx, off, 32));
            float mn = fmaxf(mrow[r], mx);
            float alpha = __expf(mrow[r] - mn);
            mrow[r] = mn;
            float p0 = __expf(a0 - mn), p1 = __expf(a1 - mn);
            float rs = p0 + p1;
#pragma unroll
            for (int off = 8; off >= 1; off >>= 1) rs += __shfl_xor(rs, off, 32);
            lrow[r] = lrow[r] * alpha + rs;
#pragma unroll
            for (int j = 0; j < 4; j++) o[j][r] *= alpha;
            int pr = r + 8 * half;
            sP[w][pr * 32 + m]      = (f16)p0;
            sP[w][pr * 32 + 16 + m] = (f16)p1;
        }
        __syncthreads();

        // O += P(16x32) @ V_tile(32x64)   (B-frags from V^T rows)
        Frag16 pf;
        {
            const f16* p = &sP[w][m * 32 + half * 8];
            pf.u[0] = *(const u32x4*)p;
            pf.u[1] = *(const u32x4*)(p + 16);
        }
#pragma unroll
        for (int j = 0; j < 4; j++) {
            Frag16 bv;
            const f16* p = &sVt[(j * 16 + m) * 32 + half * 8];
            bv.u[0] = *(const u32x4*)p; bv.u[1] = *(const u32x4*)(p + 16);
            o[j] = __builtin_amdgcn_wmma_f32_16x16x32_f16(false, pf.h, false, bv.h, (short)0, o[j], false, false);
        }
    }

#pragma unroll
    for (int r = 0; r < 8; r++) {
        float inv = 1.0f / lrow[r];
        int srow = qt * 64 + w * 16 + r + 8 * half;
        size_t orow = ((size_t)b * SEQ + srow) * HIDDEN + hh * 64;
#pragma unroll
        for (int j = 0; j < 4; j++)
            attn[orow + j * 16 + m] = (f16)(o[j][r] * inv);
    }
}

// ---------------------------------------------------------------------------
extern "C" void kernel_launch(void* const* d_in, const int* in_sizes, int n_in,
                              void* d_out, int out_size, void* d_ws, size_t ws_size,
                              hipStream_t stream)
{
    (void)in_sizes; (void)n_in; (void)out_size; (void)ws_size;
    const float* x    = (const float*)d_in[0];
    const float* qbw  = (const float*)d_in[1];
    const float* qsw  = (const float*)d_in[2];
    const float* qsc  = (const float*)d_in[3];
    const float* kbw  = (const float*)d_in[4];
    const float* ksw  = (const float*)d_in[5];
    const float* ksc  = (const float*)d_in[6];
    const float* vbw  = (const float*)d_in[7];
    const float* vsw  = (const float*)d_in[8];
    const float* vsc  = (const float*)d_in[9];
    const float* outw = (const float*)d_in[10];
    const float* outb = (const float*)d_in[11];

    char* ws = (char*)d_ws;
    size_t off = 0;
    f16* Acat = (f16*)(ws + off); off += (size_t)NTOK * KCAT * 2;
    f16* Wq   = (f16*)(ws + off); off += (size_t)HIDDEN * KCAT * 2;
    f16* Wk   = (f16*)(ws + off); off += (size_t)HIDDEN * KCAT * 2;
    f16* Wv   = (f16*)(ws + off); off += (size_t)HIDDEN * KCAT * 2;
    f16* Wout = (f16*)(ws + off); off += (size_t)HIDDEN * HIDDEN * 2;
    float* qf = (float*)(ws + off); off += (size_t)NTOK * HIDDEN * 4;
    float* kf = (float*)(ws + off); off += (size_t)NTOK * HIDDEN * 4;
    float* vf = (float*)(ws + off); off += (size_t)NTOK * HIDDEN * 4;
    f16* qh   = (f16*)(ws + off); off += (size_t)NTOK * HIDDEN * 2;
    f16* kh   = (f16*)(ws + off); off += (size_t)NTOK * HIDDEN * 2;
    f16* vh   = (f16*)(ws + off); off += (size_t)NTOK * HIDDEN * 2;
    f16* attn = (f16*)(ws + off); off += (size_t)NTOK * HIDDEN * 2;

    const int WEL = HIDDEN * HIDDEN;          // 589824
    kan_prep_weights<<<dim3(WEL / 256, 3), 256, 0, stream>>>(
        qbw, qsw, qsc, kbw, ksw, ksc, vbw, vsw, vsc, Wq, Wk, Wv);
    convert_w_f16<<<WEL / 256, 256, 0, stream>>>(outw, Wout, WEL);
    kan_prep_acts<<<(NTOK * HIDDEN) / 256, 256, 0, stream>>>(x, Acat);

    dim3 ggrid(NTOK / 64, HIDDEN / 256);      // (64, 3)
    gemm_nt_f16f32<<<ggrid, 256, 0, stream>>>(Acat, Wq, qf, nullptr, NTOK, HIDDEN, KCAT);
    gemm_nt_f16f32<<<ggrid, 256, 0, stream>>>(Acat, Wk, kf, nullptr, NTOK, HIDDEN, KCAT);
    gemm_nt_f16f32<<<ggrid, 256, 0, stream>>>(Acat, Wv, vf, nullptr, NTOK, HIDDEN, KCAT);

    rope_convert<<<(NTOK * HIDDEN) / 256, 256, 0, stream>>>(qf, kf, vf, qh, kh, vh);

    flash_attn<<<4 * HEADS * (SEQ / 64), 128, 0, stream>>>(qh, kh, vh, attn);

    gemm_nt_f16f32<<<ggrid, 256, 0, stream>>>(attn, Wout, (float*)d_out, outb,
                                              NTOK, HIDDEN, HIDDEN);
}
